// anim_conv_36575941493547
// MI455X (gfx1250) — matrix operand order repaired and checked by reading, not compiled, not run on hardware
//
#include <hip/hip_runtime.h>
#include <hip/hip_bf16.h>

// ---------------------------------------------------------------------------
// MI455X (gfx1250) implementation.
//  - Heavy GEMMs (GRU propagation + final per-channel einsum) use
//    v_wmma_f32_16x16x32_f16 (f16 in, f32 accumulate).
//  - Whole 3-step GRU fused per (n,t) slice inside LDS; Wadj (128KB), H slice
//    and GRU weights staged into LDS with the Tensor Data Mover
//    (tensor_load_to_lds + s_wait_tensorcnt), ~280KB of the 320KB WGP LDS.
//  - Phases 1/2 use 4 independent WMMA accumulator chains per wave to hide the
//    5-slot WMMA->WMMA RAW hazard (ISA 7.12.1) instead of compiler v_nops.
//  - MLPs / scatters are cheap VALU kernels.
// ---------------------------------------------------------------------------

typedef __attribute__((ext_vector_type(16))) _Float16 v16h;
typedef __attribute__((ext_vector_type(8)))  float    v8f;

#define NN   32
#define TT   64
#define VV   256
#define DDim 42
#define CC   64
#define EE   8192
#define LL   3
#define NT   (NN * TT)        // 2048
#define ROWS (NT * VV)        // 524288

// LDS layout for k_gru (in halves)
#define LDS_BUFX   0          // 16384 : H ping
#define LDS_BUFY   16384      // 16384 : H pong / M^T scratch
#define LDS_AG     32768      // 16384 : AGG
#define LDS_WL     49152      // 4096  : Wl^T (per step)
#define LDS_WIH    53248      // 12288 : Wih^T
#define LDS_WHH    65536      // 12288 : Whh^T (contiguous after Wih^T)
#define LDS_WADJ   77824      // 65536 : Wadj f16 (TDM-staged)
#define LDS_HALVES 143360     // total -> 286720 bytes

__device__ __forceinline__ float sigmoidf_(float x) { return 1.0f / (1.0f + __expf(-x)); }

// A fragment (16x32, f16) from row-major [rows, ld] storage.
__device__ __forceinline__ v16h load_a16x32(const _Float16* base, int row0, int k0, int ld) {
    const int lane = threadIdx.x & 31;
    const int m    = lane & 15;
    const int kb   = (lane >> 4) * 16;
    const _Float16* p = base + (size_t)(row0 + m) * ld + k0 + kb;
    v16h a;
#pragma unroll
    for (int j = 0; j < 8; ++j) a[j] = p[j];
#pragma unroll
    for (int j = 0; j < 8; ++j) a[8 + j] = p[8 + j];
    return a;
}

// B fragment (32x16, f16) from N-major (transposed) storage Bt[n][k].
__device__ __forceinline__ v16h load_bT32x16(const _Float16* bt, int n0, int k0, int ld) {
    const int lane = threadIdx.x & 31;
    const int n    = lane & 15;
    const int kb   = (lane >> 4) * 16;
    const _Float16* p = bt + (size_t)(n0 + n) * ld + k0 + kb;
    v16h b;
#pragma unroll
    for (int j = 0; j < 16; ++j) b[j] = p[j];
    return b;
}

__device__ __forceinline__ v8f wmma_f16(v16h a, v16h b, v8f c) {
    return __builtin_amdgcn_wmma_f32_16x16x32_f16(false, a, false, b, (short)0, c, false, false);
}

// ---------------------------------------------------------------------------
// Tensor Data Mover staging (gfx1250 TDM). 2-byte elements, 2D tile == tensor.
#if defined(__has_builtin)
#if __has_builtin(__builtin_amdgcn_tensor_load_to_lds) && __has_builtin(__builtin_amdgcn_s_wait_tensorcnt)
#define HAVE_TDM 1
#endif
#endif

#ifdef HAVE_TDM
typedef __attribute__((ext_vector_type(4))) unsigned int v4u;
typedef __attribute__((ext_vector_type(8))) int          v8i;
typedef __attribute__((ext_vector_type(4))) int          v4i;

__device__ __forceinline__ void tdm_load_f16(const void* gsrc, unsigned lds_byte_off,
                                             unsigned dim0, unsigned dim1, unsigned stride0) {
    const unsigned long long ga = (unsigned long long)(size_t)gsrc;
    v4u g0;
    g0[0] = 1u;                                                  // count=1, user-mode D#
    g0[1] = lds_byte_off;                                        // lds_addr (bytes)
    g0[2] = (unsigned)ga;                                        // global_addr[31:0]
    g0[3] = (unsigned)((ga >> 32) & 0x01ffffffu) | (2u << 30);   // addr[56:32] | type=2
    v8i g1;
    g1[0] = 0x00010000;                                          // wg_mask=0, data_size=1 (2B)
    g1[1] = (int)((dim0 & 0xffffu) << 16);                       // tensor_dim0[15:0]
    g1[2] = (int)((dim0 >> 16) | ((dim1 & 0xffffu) << 16));      // dim0 hi | dim1 lo
    g1[3] = (int)((dim1 >> 16) | ((dim0 & 0xffffu) << 16));      // dim1 hi | tile_dim0
    g1[4] = (int)(dim1 & 0xffffu);                               // tile_dim1 (tile_dim2=0)
    g1[5] = (int)stride0;                                        // tensor_dim0_stride lo
    g1[6] = 0;
    g1[7] = 0;
    v4i gz = {0, 0, 0, 0};
#if __clang_major__ >= 23
    v8i gz8 = {0, 0, 0, 0, 0, 0, 0, 0};
    __builtin_amdgcn_tensor_load_to_lds(g0, g1, gz, gz, gz8, 0);
#else
    __builtin_amdgcn_tensor_load_to_lds(g0, g1, gz, gz, 0);
#endif
}
#endif

// ---------------------------------------------------------------------------
__global__ void k_zero(float* __restrict__ p, int n) {
    int i = blockIdx.x * blockDim.x + threadIdx.x;
    if (i < n) p[i] = 0.0f;
}

// ---------------------------------------------------------------------------
// Edge MLP: [E,42] -> 42 -> 64 (Linear+BN+ReLU twice), plus scalar mean weight.
__global__ __launch_bounds__(256) void k_edge_mlp(
    const float* __restrict__ ea,
    const float* __restrict__ W1, const float* __restrict__ b1,
    const float* __restrict__ g1, const float* __restrict__ t1,
    const float* __restrict__ W2, const float* __restrict__ b2,
    const float* __restrict__ g2, const float* __restrict__ t2,
    float* __restrict__ ef, float* __restrict__ wE)
{
    __shared__ float sW1[DDim * DDim], sW2[DDim * CC];
    __shared__ float sb1[DDim], ss1[DDim], st1[DDim];
    __shared__ float sb2[CC],  ss2[CC],  st2[CC];
    const float rs = rsqrtf(1.0f + 1e-5f);
    for (int i = threadIdx.x; i < DDim * DDim; i += 256) sW1[i] = W1[i];
    for (int i = threadIdx.x; i < DDim * CC;   i += 256) sW2[i] = W2[i];
    for (int i = threadIdx.x; i < DDim; i += 256) { sb1[i] = b1[i]; ss1[i] = g1[i] * rs; st1[i] = t1[i]; }
    for (int i = threadIdx.x; i < CC;   i += 256) { sb2[i] = b2[i]; ss2[i] = g2[i] * rs; st2[i] = t2[i]; }
    __syncthreads();

    int e = blockIdx.x * 256 + threadIdx.x;
    if (e >= EE) return;
    float xin[DDim];
#pragma unroll
    for (int i = 0; i < DDim; ++i) xin[i] = ea[(size_t)e * DDim + i];
    float acc[CC];
#pragma unroll
    for (int j = 0; j < CC; ++j) acc[j] = sb2[j];
    for (int j1 = 0; j1 < DDim; ++j1) {
        float a = sb1[j1];
#pragma unroll
        for (int i = 0; i < DDim; ++i) a += xin[i] * sW1[i * DDim + j1];
        a = fmaxf(a * ss1[j1] + st1[j1], 0.0f);
#pragma unroll
        for (int j = 0; j < CC; ++j) acc[j] += a * sW2[j1 * CC + j];
    }
    float s = 0.0f;
#pragma unroll
    for (int j = 0; j < CC; ++j) {
        float v = fmaxf(acc[j] * ss2[j] + st2[j], 0.0f);
        ef[(size_t)e * CC + j] = v;
        s += v;
    }
    wE[e] = s * (1.0f / CC);
}

// ---------------------------------------------------------------------------
// Node MLP: [524288,42] -> 42 -> 64; stores h as f16 (WMMA operand precision).
__global__ __launch_bounds__(256) void k_node_mlp(
    const float* __restrict__ x,
    const float* __restrict__ W1, const float* __restrict__ b1,
    const float* __restrict__ g1, const float* __restrict__ t1,
    const float* __restrict__ W2, const float* __restrict__ b2,
    const float* __restrict__ g2, const float* __restrict__ t2,
    _Float16* __restrict__ hF)
{
    __shared__ float sW1[DDim * DDim], sW2[DDim * CC];
    __shared__ float sb1[DDim], ss1[DDim], st1[DDim];
    __shared__ float sb2[CC],  ss2[CC],  st2[CC];
    const float rs = rsqrtf(1.0f + 1e-5f);
    for (int i = threadIdx.x; i < DDim * DDim; i += 256) sW1[i] = W1[i];
    for (int i = threadIdx.x; i < DDim * CC;   i += 256) sW2[i] = W2[i];
    for (int i = threadIdx.x; i < DDim; i += 256) { sb1[i] = b1[i]; ss1[i] = g1[i] * rs; st1[i] = t1[i]; }
    for (int i = threadIdx.x; i < CC;   i += 256) { sb2[i] = b2[i]; ss2[i] = g2[i] * rs; st2[i] = t2[i]; }
    __syncthreads();

    int r = blockIdx.x * 256 + threadIdx.x;
    if (r >= ROWS) return;
    float xin[DDim];
#pragma unroll
    for (int i = 0; i < DDim; ++i) xin[i] = x[(size_t)r * DDim + i];
    float acc[CC];
#pragma unroll
    for (int j = 0; j < CC; ++j) acc[j] = sb2[j];
    for (int j1 = 0; j1 < DDim; ++j1) {
        float a = sb1[j1];
#pragma unroll
        for (int i = 0; i < DDim; ++i) a += xin[i] * sW1[i * DDim + j1];
        a = fmaxf(a * ss1[j1] + st1[j1], 0.0f);
#pragma unroll
        for (int j = 0; j < CC; ++j) acc[j] += a * sW2[j1 * CC + j];
    }
#pragma unroll
    for (int j = 0; j < CC; ++j)
        hF[(size_t)r * CC + j] = (_Float16)fmaxf(acc[j] * ss2[j] + st2[j], 0.0f);
}

// ---------------------------------------------------------------------------
// Scatter: Wadj[dst][src] += mean(ef);  AT[c][dst][src] += ef[e][c]
__global__ void k_scatter(const float* __restrict__ ef, const float* __restrict__ wE,
                          const int* __restrict__ idx,
                          float* __restrict__ Wadj, float* __restrict__ AT)
{
    int t = blockIdx.x * blockDim.x + threadIdx.x;
    if (t >= EE * CC) return;
    int e = t >> 6, c = t & 63;
    int src = idx[e], dst = idx[EE + e];
    atomicAdd(&AT[(size_t)c * (VV * VV) + dst * VV + src], ef[(size_t)e * CC + c]);
    if (c == 0) atomicAdd(&Wadj[dst * VV + src], wE[e]);
}

// ---------------------------------------------------------------------------
// Prep: f16 adjacency + transposed f16 GRU weights (N-major for B-fragments).
__global__ void k_prep(const float* __restrict__ Wadj,
                       const float* __restrict__ Wl, const float* __restrict__ Wih,
                       const float* __restrict__ Whh,
                       _Float16* __restrict__ WadjH, _Float16* __restrict__ WlT,
                       _Float16* __restrict__ WihT,  _Float16* __restrict__ WhhT)
{
    int t = blockIdx.x * blockDim.x + threadIdx.x;
    if (t < VV * VV) WadjH[t] = (_Float16)Wadj[t];
    if (t < LL * CC * CC) {
        int l = t >> 12, r = t & 4095, n = r >> 6, k = r & 63;
        WlT[t] = (_Float16)Wl[l * 4096 + k * 64 + n];
    }
    if (t < 192 * 64) {
        int n = t >> 6, k = t & 63;
        WihT[t] = (_Float16)Wih[k * 192 + n];
        WhhT[t] = (_Float16)Whh[k * 192 + n];
    }
}

// ---------------------------------------------------------------------------
// Fused GatedGraphConv: one workgroup (8 waves) per (n,t) slice.
__global__ __launch_bounds__(256) void k_gru(
    _Float16* __restrict__ hF, const _Float16* __restrict__ WadjH,
    const _Float16* __restrict__ WlTg, const _Float16* __restrict__ WihTg,
    const _Float16* __restrict__ WhhTg,
    const float* __restrict__ bih, const float* __restrict__ bhh)
{
    extern __shared__ _Float16 sm[];
    _Float16* bufX  = sm + LDS_BUFX;
    _Float16* bufY  = sm + LDS_BUFY;
    _Float16* AG    = sm + LDS_AG;
    _Float16* sWl   = sm + LDS_WL;
    _Float16* sWih  = sm + LDS_WIH;
    _Float16* sWhh  = sm + LDS_WHH;
    _Float16* sWadj = sm + LDS_WADJ;

    const int nt   = blockIdx.x;
    const int tid  = threadIdx.x;
    const int wave = tid >> 5;
    const int lane = tid & 31;
    const int ln   = lane & 15;
    const int lm   = (lane >> 4) * 8;
    const size_t base = (size_t)nt * (VV * CC);

    _Float16* Hold = bufX;
    _Float16* Hnew = bufY;

#ifdef HAVE_TDM
    // Stage H slice, GRU weights (Wih^T|Whh^T contiguous) and Wadj via the TDM.
    if (tid < 32) {
        tdm_load_f16(hF + base, LDS_BUFX * 2u, VV * CC, 1u, VV * CC);
        tdm_load_f16(WihTg,     LDS_WIH  * 2u, 2u * 192u * 64u, 1u, 2u * 192u * 64u);
        tdm_load_f16(WadjH,     LDS_WADJ * 2u, VV, VV, VV);
        __builtin_amdgcn_s_wait_tensorcnt(0);
    }
#else
#pragma unroll
    for (int i = 0; i < 64; ++i) Hold[tid + 256 * i] = hF[base + tid + 256 * i];
    for (int i = tid; i < 192 * 64; i += 256) { sWih[i] = WihTg[i]; sWhh[i] = WhhTg[i]; }
    for (int i = tid; i < VV * VV; i += 256) sWadj[i] = WadjH[i];
#endif

    for (int l = 0; l < LL; ++l) {
        __syncthreads();
        for (int i = tid; i < 4096; i += 256) sWl[i] = WlTg[l * 4096 + i];
        __syncthreads();

        // Phase 1: M^T = (Hold @ Wl)^T stored c-major into Hnew region.
        // Two M-tiles per wave, 4 independent accumulator chains (hazard hiding).
#pragma unroll
        for (int mi = 0; mi < 2; ++mi) {
            const int mt = wave * 2 + mi;
            v8f acc[4] = {};
#pragma unroll
            for (int kk = 0; kk < 2; ++kk) {
                v16h a = load_a16x32(Hold, mt * 16, kk * 32, CC);
#pragma unroll
                for (int ct = 0; ct < 4; ++ct)
                    acc[ct] = wmma_f16(a, load_bT32x16(sWl, ct * 16, kk * 32, CC), acc[ct]);
            }
            const int mbase = mt * 16 + lm;
#pragma unroll
            for (int ct = 0; ct < 4; ++ct) {
                const int cidx = ct * 16 + ln;
#pragma unroll
                for (int i = 0; i < 8; ++i) Hnew[cidx * VV + mbase + i] = (_Float16)acc[ct][i];
            }
        }
        __syncthreads();

        // Phase 2: AGG = Wadj @ M (Wadj A-tiles from LDS; shared A, 4 chains).
#pragma unroll
        for (int mi = 0; mi < 2; ++mi) {
            const int mt = wave * 2 + mi;
            v8f acc[4] = {};
#pragma unroll
            for (int kt = 0; kt < 8; ++kt) {
                v16h a = load_a16x32(sWadj, mt * 16, kt * 32, VV);
#pragma unroll
                for (int ct = 0; ct < 4; ++ct)
                    acc[ct] = wmma_f16(a, load_bT32x16(Hnew, ct * 16, kt * 32, VV), acc[ct]);
            }
            const int mbase = mt * 16 + lm;
#pragma unroll
            for (int ct = 0; ct < 4; ++ct) {
                const int cidx = ct * 16 + ln;
#pragma unroll
                for (int i = 0; i < 8; ++i) AG[(mbase + i) * CC + cidx] = (_Float16)acc[ct][i];
            }
        }
        __syncthreads();

        // Phase 3: GI = AGG@Wih, GH = Hold@Whh (r/z/n slices), GRU gates.
        for (int tile = wave; tile < 64; tile += 8) {
            int mt = tile >> 2, ct = tile & 3;
            v8f gir = {}, giz = {}, gin = {}, ghr = {}, ghz = {}, ghn = {};
#pragma unroll
            for (int kk = 0; kk < 2; ++kk) {
                int k0 = kk * 32;
                v16h aA = load_a16x32(AG,   mt * 16, k0, CC);
                v16h aH = load_a16x32(Hold, mt * 16, k0, CC);
                gir = wmma_f16(aA, load_bT32x16(sWih,       ct * 16, k0, CC), gir);
                giz = wmma_f16(aA, load_bT32x16(sWih,  64 + ct * 16, k0, CC), giz);
                gin = wmma_f16(aA, load_bT32x16(sWih, 128 + ct * 16, k0, CC), gin);
                ghr = wmma_f16(aH, load_bT32x16(sWhh,       ct * 16, k0, CC), ghr);
                ghz = wmma_f16(aH, load_bT32x16(sWhh,  64 + ct * 16, k0, CC), ghz);
                ghn = wmma_f16(aH, load_bT32x16(sWhh, 128 + ct * 16, k0, CC), ghn);
            }
            int col = ct * 16 + ln;
            float bir = bih[col],       bhr = bhh[col];
            float biz = bih[64 + col],  bhz = bhh[64 + col];
            float bin = bih[128 + col], bhn = bhh[128 + col];
            int mbase = mt * 16 + lm;
#pragma unroll
            for (int i = 0; i < 8; ++i) {
                float ho   = (float)Hold[(mbase + i) * CC + col];
                float r    = sigmoidf_(gir[i] + bir + ghr[i] + bhr);
                float z    = sigmoidf_(giz[i] + biz + ghz[i] + bhz);
                float cand = tanhf((gin[i] + bin) + r * (ghn[i] + bhn));
                Hnew[(mbase + i) * CC + col] = (_Float16)((1.0f - z) * cand + z * ho);
            }
        }
        _Float16* tmp = Hold; Hold = Hnew; Hnew = tmp;
    }
    __syncthreads();
    // final ReLU, write back (in-place per slice; re-derived from inputs each launch)
#pragma unroll
    for (int i = 0; i < 64; ++i) {
        _Float16 v = Hold[tid + 256 * i];
        hF[base + tid + 256 * i] = v > (_Float16)0 ? v : (_Float16)0;
    }
}

// ---------------------------------------------------------------------------
// Final einsum: out[n,c,t,w] = sum_v h[n,t,v,c] * AT[c][w][v].
__global__ __launch_bounds__(256) void k_out(
    const _Float16* __restrict__ hF, const float* __restrict__ AT,
    float* __restrict__ out)
{
    const int gwave = (blockIdx.x * 256 + threadIdx.x) >> 5;
    const int lane  = threadIdx.x & 31;
    const int c     = gwave >> 7;    // 0..63
    const int mt    = gwave & 127;   // 0..127
    const int row0  = mt * 16;
    const int ln    = lane & 15;
    const int lm    = (lane >> 4) * 8;

    v16h afr[8];
#pragma unroll
    for (int kt = 0; kt < 8; ++kt) {
        const int row = row0 + ln;
        const int kb  = kt * 32 + lm;
        v16h a;
#pragma unroll
        for (int j = 0; j < 8; ++j) a[j]     = hF[(size_t)row * (VV * CC) + (size_t)(kb + lm + j) * CC + c];
#pragma unroll
        for (int j = 0; j < 8; ++j) a[8 + j] = hF[(size_t)row * (VV * CC) + (size_t)(kb + lm + 8 + j) * CC + c];
        afr[kt] = a;
    }
    const float* ATc = AT + (size_t)c * (VV * VV);
    const int kbB = (lane >> 4) * 16;
    for (int wt = 0; wt < 16; ++wt) {
        if (wt < 15)
            __builtin_prefetch(ATc + (size_t)((wt + 1) * 16 + ln) * VV + kbB, 0, 0);
        v8f acc = {};
#pragma unroll
        for (int kt = 0; kt < 8; ++kt) {
            const float* p = ATc + (size_t)(wt * 16 + ln) * VV + kt * 32 + kbB;
            v16h b;
#pragma unroll
            for (int j = 0; j < 16; ++j) b[j] = (_Float16)p[j];
            acc = wmma_f16(afr[kt], b, acc);
        }
        const int w = wt * 16 + ln;
#pragma unroll
        for (int i = 0; i < 8; ++i) {
            int row = row0 + lm + i;
            int n_ = row >> 6, t_ = row & 63;
            out[(size_t)((n_ * CC + c) * TT + t_) * VV + w] = acc[i];
        }
    }
}

// ---------------------------------------------------------------------------
extern "C" void kernel_launch(void* const* d_in, const int* in_sizes, int n_in,
                              void* d_out, int out_size, void* d_ws, size_t ws_size,
                              hipStream_t stream)
{
    const float* x    = (const float*)d_in[0];
    const float* ea   = (const float*)d_in[1];
    const int*   eidx = (const int*)  d_in[2];
    const float* eW1  = (const float*)d_in[3];
    const float* eb1  = (const float*)d_in[4];
    const float* eg1  = (const float*)d_in[5];
    const float* et1  = (const float*)d_in[6];
    const float* eW2  = (const float*)d_in[7];
    const float* eb2  = (const float*)d_in[8];
    const float* eg2  = (const float*)d_in[9];
    const float* et2  = (const float*)d_in[10];
    const float* nW1  = (const float*)d_in[11];
    const float* nb1  = (const float*)d_in[12];
    const float* ng1  = (const float*)d_in[13];
    const float* nt1  = (const float*)d_in[14];
    const float* nW2  = (const float*)d_in[15];
    const float* nb2  = (const float*)d_in[16];
    const float* ng2  = (const float*)d_in[17];
    const float* nt2  = (const float*)d_in[18];
    const float* Wl   = (const float*)d_in[19];
    const float* Wih  = (const float*)d_in[20];
    const float* Whh  = (const float*)d_in[21];
    const float* bih  = (const float*)d_in[22];
    const float* bhh  = (const float*)d_in[23];

    // Workspace carve (Wadj and AT kept adjacent so one zero kernel covers both).
    char* ws = (char*)d_ws;
    float* ef   = (float*)ws;  ws += sizeof(float) * (size_t)EE * CC;
    float* wE   = (float*)ws;  ws += sizeof(float) * (size_t)EE;
    float* Wadj = (float*)ws;  ws += sizeof(float) * (size_t)VV * VV;
    float* AT   = (float*)ws;  ws += sizeof(float) * (size_t)CC * VV * VV;
    _Float16* WadjH = (_Float16*)ws; ws += sizeof(_Float16) * (size_t)VV * VV;
    _Float16* WlT   = (_Float16*)ws; ws += sizeof(_Float16) * (size_t)LL * CC * CC;
    _Float16* WihT  = (_Float16*)ws; ws += sizeof(_Float16) * (size_t)192 * 64;
    _Float16* WhhT  = (_Float16*)ws; ws += sizeof(_Float16) * (size_t)192 * 64;
    _Float16* hF    = (_Float16*)ws; ws += sizeof(_Float16) * (size_t)ROWS * CC;

    const int nzero = VV * VV + CC * VV * VV;      // Wadj + AT
    k_zero<<<(nzero + 255) / 256, 256, 0, stream>>>(Wadj, nzero);

    k_edge_mlp<<<EE / 256, 256, 0, stream>>>(ea, eW1, eb1, eg1, et1, eW2, eb2, eg2, et2, ef, wE);
    k_node_mlp<<<ROWS / 256, 256, 0, stream>>>(x, nW1, nb1, ng1, nt1, nW2, nb2, ng2, nt2, hF);
    k_scatter<<<(EE * CC) / 256, 256, 0, stream>>>(ef, wE, eidx, Wadj, AT);
    k_prep<<<(VV * VV) / 256, 256, 0, stream>>>(Wadj, Wl, Wih, Whh, WadjH, WlT, WihT, WhhT);

    const size_t gru_lds = (size_t)LDS_HALVES * sizeof(_Float16);  // 280KB dynamic LDS
    k_gru<<<NT, 256, gru_lds, stream>>>(hF, WadjH, WlT, WihT, WhhT, bih, bhh);

    k_out<<<(64 * 128) / 8, 256, 0, stream>>>(hF, AT, (float*)d_out);
}
